// AttentionHead_19112604467527
// MI455X (gfx1250) — compile-verified, matrix operand order checked by reading, not run on hardware
//
#include <hip/hip_runtime.h>

// Shapes (fixed by the reference)
#define BB   4
#define NN   2048
#define DD   1024
#define HH   1024
#define MT   (BB * NN)          // 8192 total rows of x / Q / K / V

typedef __attribute__((ext_vector_type(16))) __bf16 bf16x16;
typedef __attribute__((ext_vector_type(8)))  float  f32x8;

union Frag { bf16x16 v; uint4 q[2]; };

__device__ __forceinline__ unsigned short f2bf(float x) {
  unsigned int u = __float_as_uint(x);
  u += 0x7FFFu + ((u >> 16) & 1u);      // round-to-nearest-even
  return (unsigned short)(u >> 16);
}

// ---------------------------------------------------------------------------
// CDNA5 async DMA: global -> LDS, 16B per lane, tracked by ASYNCcnt.
// ---------------------------------------------------------------------------
__device__ __forceinline__ void async_b128(unsigned lds_off, unsigned long long g) {
  asm volatile("global_load_async_to_lds_b128 %0, %1, off"
               :: "v"(lds_off), "v"(g) : "memory");
}
__device__ __forceinline__ void wait_async0() {
  asm volatile("s_wait_asynccnt 0" ::: "memory");
}

// Stage ROWS x 32 bf16 (ROWS x 64 bytes) into LDS at ldsbase.
// LDS layout: row r occupies bytes [r*64, r*64+64)  (K-major within row).
// Chunks of 16B are distributed across the 8 waves / 32 lanes.
template<int ROWS>
__device__ __forceinline__ void stage(const unsigned short* gbase, size_t ld,
                                      int row0, int k0, unsigned ldsbase,
                                      int wave, int lane) {
  constexpr int CHUNKS   = ROWS * 4;        // 16B chunks
  constexpr int PER_WAVE = CHUNKS / 8;
  constexpr int ITERS    = PER_WAVE / 32;
#pragma unroll
  for (int it = 0; it < ITERS; ++it) {
    int c   = wave * PER_WAVE + it * 32 + lane;
    int r   = c >> 2;
    int seg = c & 3;
    unsigned long long g =
        (unsigned long long)(gbase + (size_t)(row0 + r) * ld + (size_t)k0 + seg * 8);
    async_b128(ldsbase + (unsigned)c * 16u, g);
  }
}

// Load one 16x32 bf16 fragment slice from an LDS row base.
// p = rowbase + hl*16 ; lanes 0-15 read bytes {0..15, 32..47} of the 64B row
// (K 0..7,16..23), lanes 16-31 read {16..31, 48..63} (K 8..15,24..31).
__device__ __forceinline__ Frag ldfrag_lds(const char* p) {
  Frag f;
  f.q[0] = *(const uint4*)(p);
  f.q[1] = *(const uint4*)(p + 32);
  return f;
}

// Shared GEMM core: acc[4][4] (64x64 per wave), workgroup tile 128x256,
// double-buffered async staging. A rows along M, B rows along N (both K-major).
__device__ __forceinline__ void gemm_core(const unsigned short* __restrict__ Abase, size_t ldA, int am0,
                                          const unsigned short* __restrict__ Bbase, size_t ldB, int bn0,
                                          int K, char* smem, int wave, int lane,
                                          int wm, int wn, int hl, int ln,
                                          f32x8 acc[4][4]) {
  const unsigned BUF[2] = {0u, 24576u};     // A: 8KB, B: 16KB per buffer

  // Prologue: stage k=0 into buffer 0.
  stage<128>(Abase, ldA, am0, 0, BUF[0] + 0u,    wave, lane);
  stage<256>(Bbase, ldB, bn0, 0, BUF[0] + 8192u, wave, lane);
  wait_async0();
  __syncthreads();

  int cur = 0;
  for (int k0 = 0; k0 < K; k0 += 32) {
    if (k0 + 32 < K) {   // stage next slab while computing this one
      stage<128>(Abase, ldA, am0, k0 + 32, BUF[cur ^ 1] + 0u,    wave, lane);
      stage<256>(Bbase, ldB, bn0, k0 + 32, BUF[cur ^ 1] + 8192u, wave, lane);
    }

    const char* As = smem + BUF[cur];
    const char* Bs = smem + BUF[cur] + 8192;
    Frag a[4], b4[4];
#pragma unroll
    for (int i = 0; i < 4; ++i)
      a[i]  = ldfrag_lds(As + (size_t)(wm * 64 + 16 * i + ln) * 64 + hl * 16);
#pragma unroll
    for (int j = 0; j < 4; ++j)
      b4[j] = ldfrag_lds(Bs + (size_t)(wn * 64 + 16 * j + ln) * 64 + hl * 16);
#pragma unroll
    for (int i = 0; i < 4; ++i)
#pragma unroll
      for (int j = 0; j < 4; ++j)
        acc[i][j] = __builtin_amdgcn_wmma_f32_16x16x32_bf16(
            false, a[i].v, false, b4[j].v, (short)0, acc[i][j], false, false);

    wait_async0();       // next slab landed
    __syncthreads();     // everyone done reading current slab
    cur ^= 1;
  }
}

__global__ void f32_to_bf16_kernel(const float* __restrict__ s,
                                   unsigned short* __restrict__ d, int n) {
  int i = blockIdx.x * blockDim.x + threadIdx.x;
  if (i < n) d[i] = f2bf(s[i]);
}

// ---------------------------------------------------------------------------
// Q/K/V projection: C[8192,1024] = Xb[8192,1024] @ W[1024,1024]^T  (3 matrices)
// Workgroup tile 128x256; V stored transposed for the final GEMM.
// ---------------------------------------------------------------------------
__global__ __launch_bounds__(256)
void qkv_gemm(const unsigned short* __restrict__ xb,
              const unsigned short* __restrict__ wqb,
              const unsigned short* __restrict__ wkb,
              const unsigned short* __restrict__ wvb,
              unsigned short* __restrict__ Qb,
              unsigned short* __restrict__ Kb,
              unsigned short* __restrict__ Vt) {
  extern __shared__ char smem[];
  const int lane = threadIdx.x & 31;
  const int wave = threadIdx.x >> 5;
  const int hl   = lane >> 4;
  const int ln   = lane & 15;
  const int wm   = wave >> 2;       // 0..1
  const int wn   = wave & 3;        // 0..3

  const int id  = blockIdx.x;       // 0..767
  const int mat = id >> 8;          // /256 : 0=Q 1=K 2=V
  const int t   = id & 255;
  const int m0  = (t >> 2) * 128;   // 64 row tiles
  const int n0  = (t & 3) * 256;    // 4 col tiles
  const unsigned short* W = (mat == 0) ? wqb : (mat == 1) ? wkb : wvb;

  f32x8 acc[4][4];
  const f32x8 z = {0.f,0.f,0.f,0.f,0.f,0.f,0.f,0.f};
#pragma unroll
  for (int i = 0; i < 4; ++i)
#pragma unroll
    for (int j = 0; j < 4; ++j) acc[i][j] = z;

  gemm_core(xb, DD, m0, W, DD, n0, DD, smem, wave, lane, wm, wn, hl, ln, acc);

  // C/D layout: VGPR v -> M = v + 8*(lane>=16); N = lane&15
  const int mw = m0 + wm * 64, nw = n0 + wn * 64;
  if (mat < 2) {
    unsigned short* O = (mat == 0) ? Qb : Kb;
#pragma unroll
    for (int i = 0; i < 4; ++i)
#pragma unroll
      for (int j = 0; j < 4; ++j)
#pragma unroll
        for (int v = 0; v < 8; ++v) {
          int row = mw + 16 * i + 8 * hl + v;
          int col = nw + 16 * j + ln;
          O[(size_t)row * HH + col] = f2bf(acc[i][j][v]);
        }
  } else {
#pragma unroll
    for (int i = 0; i < 4; ++i)
#pragma unroll
      for (int j = 0; j < 4; ++j)
#pragma unroll
        for (int v = 0; v < 8; ++v) {
          int row = mw + 16 * i + 8 * hl + v;
          int col = nw + 16 * j + ln;
          Vt[(size_t)col * MT + row] = f2bf(acc[i][j][v]);   // V transposed
        }
  }
}

// ---------------------------------------------------------------------------
// scores[b] = Q[b] @ K[b]^T / 32, masked. Written fp32 into the alpha output.
// ---------------------------------------------------------------------------
__global__ __launch_bounds__(256)
void scores_gemm(const unsigned short* __restrict__ Qb,
                 const unsigned short* __restrict__ Kb,
                 const int* __restrict__ mask,
                 float* __restrict__ sc) {
  extern __shared__ char smem[];
  const int lane = threadIdx.x & 31;
  const int wave = threadIdx.x >> 5;
  const int hl   = lane >> 4;
  const int ln   = lane & 15;
  const int wm   = wave >> 2;
  const int wn   = wave & 3;

  const int id = blockIdx.x;        // 0..511
  const int b  = id >> 7;           // /128
  const int t  = id & 127;
  const int q0 = (t >> 3) * 128;    // 16 row tiles
  const int c0 = (t & 7) * 256;     // 8 col tiles
  const unsigned short* QB = Qb + (size_t)b * NN * HH;
  const unsigned short* KB = Kb + (size_t)b * NN * HH;

  f32x8 acc[4][4];
  const f32x8 z = {0.f,0.f,0.f,0.f,0.f,0.f,0.f,0.f};
#pragma unroll
  for (int i = 0; i < 4; ++i)
#pragma unroll
    for (int j = 0; j < 4; ++j) acc[i][j] = z;

  gemm_core(QB, HH, q0, KB, HH, c0, HH, smem, wave, lane, wm, wn, hl, ln, acc);

  const float scale = 0.03125f;     // 1/sqrt(1024)
  const int qw = q0 + wm * 64, cw = c0 + wn * 64;
#pragma unroll
  for (int i = 0; i < 4; ++i)
#pragma unroll
    for (int j = 0; j < 4; ++j)
#pragma unroll
      for (int v = 0; v < 8; ++v) {
        int row = qw + 16 * i + 8 * hl + v;
        int col = cw + 16 * j + ln;
        size_t idx = ((size_t)b * NN + row) * NN + col;
        float s = acc[i][j][v] * scale;
        sc[idx] = (mask[idx] == 0) ? -__builtin_inff() : s;
      }
}

// ---------------------------------------------------------------------------
// Row softmax over 2048 cols; normalizes fp32 alpha in place + bf16 copy.
// ---------------------------------------------------------------------------
__global__ __launch_bounds__(256)
void softmax_rows(float* __restrict__ sc, unsigned short* __restrict__ aB) {
  const int row = blockIdx.x;                 // 0..8191
  float* r = sc + (size_t)row * NN;
  unsigned short* rb = aB + (size_t)row * NN;
  __shared__ float red[256];

  float v[8];
  float m = -__builtin_inff();
#pragma unroll
  for (int i = 0; i < 8; ++i) { v[i] = r[threadIdx.x + 256*i]; m = fmaxf(m, v[i]); }
  red[threadIdx.x] = m;
  __syncthreads();
  for (int s = 128; s > 0; s >>= 1) {
    if ((int)threadIdx.x < s) red[threadIdx.x] = fmaxf(red[threadIdx.x], red[threadIdx.x + s]);
    __syncthreads();
  }
  m = red[0];
  __syncthreads();

  float sum = 0.f;
#pragma unroll
  for (int i = 0; i < 8; ++i) { v[i] = __expf(v[i] - m); sum += v[i]; }
  red[threadIdx.x] = sum;
  __syncthreads();
  for (int s = 128; s > 0; s >>= 1) {
    if ((int)threadIdx.x < s) red[threadIdx.x] += red[threadIdx.x + s];
    __syncthreads();
  }
  float inv = 1.0f / red[0];

#pragma unroll
  for (int i = 0; i < 8; ++i) {
    float a = v[i] * inv;
    r[threadIdx.x + 256*i]  = a;
    rb[threadIdx.x + 256*i] = f2bf(a);
  }
}

// ---------------------------------------------------------------------------
// attn_output[b] = alpha[b] @ V[b]   (V held transposed: Vt[h][b*2048+k])
// ---------------------------------------------------------------------------
__global__ __launch_bounds__(256)
void attn_gemm(const unsigned short* __restrict__ aB,
               const unsigned short* __restrict__ Vt,
               float* __restrict__ out) {
  extern __shared__ char smem[];
  const int lane = threadIdx.x & 31;
  const int wave = threadIdx.x >> 5;
  const int hl   = lane >> 4;
  const int ln   = lane & 15;
  const int wm   = wave >> 2;
  const int wn   = wave & 3;

  const int id = blockIdx.x;        // 0..255
  const int b  = id >> 6;           // /64
  const int t  = id & 63;
  const int q0 = (t >> 2) * 128;    // 16 row tiles
  const int h0 = (t & 3) * 256;     // 4 col tiles

  const unsigned short* AB = aB + (size_t)b * NN * NN;      // ld = 2048
  const unsigned short* VB = Vt + (size_t)b * NN;           // ld = 8192, rows = h

  f32x8 acc[4][4];
  const f32x8 z = {0.f,0.f,0.f,0.f,0.f,0.f,0.f,0.f};
#pragma unroll
  for (int i = 0; i < 4; ++i)
#pragma unroll
    for (int j = 0; j < 4; ++j) acc[i][j] = z;

  gemm_core(AB, NN, q0, VB, MT, h0, NN, smem, wave, lane, wm, wn, hl, ln, acc);

  const int qw = q0 + wm * 64, hw = h0 + wn * 64;
#pragma unroll
  for (int i = 0; i < 4; ++i)
#pragma unroll
    for (int j = 0; j < 4; ++j)
#pragma unroll
      for (int v = 0; v < 8; ++v) {
        int row = qw + 16 * i + 8 * hl + v;
        int col = hw + 16 * j + ln;
        out[((size_t)b * NN + row) * HH + col] = acc[i][j][v];
      }
}

extern "C" void kernel_launch(void* const* d_in, const int* in_sizes, int n_in,
                              void* d_out, int out_size, void* d_ws, size_t ws_size,
                              hipStream_t stream) {
  (void)in_sizes; (void)n_in; (void)out_size; (void)ws_size;

  const float* x    = (const float*)d_in[0];
  const int*   mask = (const int*)d_in[1];
  const float* Wq   = (const float*)d_in[2];
  const float* Wk   = (const float*)d_in[3];
  const float* Wv   = (const float*)d_in[4];

  float* out_attn  = (float*)d_out;                       // [4,2048,1024]
  float* out_alpha = out_attn + (size_t)MT * HH;          // [4,2048,2048]

  // Workspace layout (bf16, ~107 MB total)
  unsigned short* xb  = (unsigned short*)d_ws;            // 8192*1024
  unsigned short* wqb = xb  + (size_t)MT * DD;            // 1024*1024
  unsigned short* wkb = wqb + (size_t)HH * DD;
  unsigned short* wvb = wkb + (size_t)HH * DD;
  unsigned short* Qb  = wvb + (size_t)HH * DD;            // 8192*1024
  unsigned short* Kb  = Qb  + (size_t)MT * HH;
  unsigned short* Vt  = Kb  + (size_t)MT * HH;            // 1024*8192 (transposed)
  unsigned short* aBf = Vt  + (size_t)HH * MT;            // 4*2048*2048

  const int nx = MT * DD;        // 8,388,608
  const int nw = HH * DD;        // 1,048,576
  f32_to_bf16_kernel<<<(nx + 255) / 256, 256, 0, stream>>>(x,  xb,  nx);
  f32_to_bf16_kernel<<<(nw + 255) / 256, 256, 0, stream>>>(Wq, wqb, nw);
  f32_to_bf16_kernel<<<(nw + 255) / 256, 256, 0, stream>>>(Wk, wkb, nw);
  f32_to_bf16_kernel<<<(nw + 255) / 256, 256, 0, stream>>>(Wv, wvb, nw);

  const size_t lds = 49152;      // 2 x (8KB A + 16KB B)
  qkv_gemm   <<<768, 256, lds, stream>>>(xb, wqb, wkb, wvb, Qb, Kb, Vt);
  scores_gemm<<<512, 256, lds, stream>>>(Qb, Kb, mask, out_alpha);
  softmax_rows<<<MT, 256, 0, stream>>>(out_alpha, aBf);
  attn_gemm  <<<256, 256, lds, stream>>>(aBf, Vt, out_attn);
}
